// CustomMoE_13855564497415
// MI455X (gfx1250) — matrix operand order repaired and checked by reading, not compiled, not run on hardware
//
#include <hip/hip_runtime.h>
#include <hip/hip_bf16.h>

#define NTOK 8192
#define DDIM 1024
#define HDIM 4096
#define NEXP 8
#define CAP  2048   // k*N/E

typedef __attribute__((ext_vector_type(16))) __bf16 v16bf;
typedef __attribute__((ext_vector_type(8)))  float  v8f;

union Frag16 { uint4 q[2]; v16bf v; };

__device__ __forceinline__ unsigned short f2bf(float f) {
  unsigned int u = __float_as_uint(f);
  u += 0x7FFFu + ((u >> 16) & 1u);   // round-to-nearest-even
  return (unsigned short)(u >> 16);
}

// ---------------- gate: logits, softmax, top-2, partial me/ce ----------------
__global__ void gate_kernel(const float* __restrict__ x, const float* __restrict__ wg,
                            int* __restrict__ topi, float* __restrict__ topv,
                            float* __restrict__ me_part, float* __restrict__ ce_part) {
  __shared__ float sMe[8][NEXP];
  __shared__ float sCe[8][NEXP];
  const int tid = threadIdx.x;
  const int wid = tid >> 5;
  const int lane = tid & 31;
  const int n = blockIdx.x * 8 + wid;

  float acc[NEXP];
#pragma unroll
  for (int e = 0; e < NEXP; ++e) acc[e] = 0.f;

  const float* xr = x + (size_t)n * DDIM;
#pragma unroll 4
  for (int it = 0; it < DDIM / 32; ++it) {
    int d = it * 32 + lane;
    float xv = xr[d];
    const float4* wr = reinterpret_cast<const float4*>(wg + (size_t)d * NEXP);
    float4 w0 = wr[0], w1 = wr[1];
    acc[0] += xv * w0.x; acc[1] += xv * w0.y; acc[2] += xv * w0.z; acc[3] += xv * w0.w;
    acc[4] += xv * w1.x; acc[5] += xv * w1.y; acc[6] += xv * w1.z; acc[7] += xv * w1.w;
  }
#pragma unroll
  for (int off = 16; off >= 1; off >>= 1)
#pragma unroll
    for (int e = 0; e < NEXP; ++e)
      acc[e] += __shfl_xor(acc[e], off, 32);

  // softmax (redundant across lanes)
  float mx = acc[0];
#pragma unroll
  for (int e = 1; e < NEXP; ++e) mx = fmaxf(mx, acc[e]);
  float p[NEXP], s = 0.f;
#pragma unroll
  for (int e = 0; e < NEXP; ++e) { p[e] = __expf(acc[e] - mx); s += p[e]; }
  float inv = 1.f / s;
#pragma unroll
  for (int e = 0; e < NEXP; ++e) p[e] *= inv;

  int i0 = 0; float v0 = p[0];
#pragma unroll
  for (int e = 1; e < NEXP; ++e) if (p[e] > v0) { v0 = p[e]; i0 = e; }
  int i1 = (i0 == 0) ? 1 : 0; float v1 = p[i1];
#pragma unroll
  for (int e = 0; e < NEXP; ++e) if (e != i0 && p[e] > v1) { v1 = p[e]; i1 = e; }

  if (lane == 0) {
    topi[n * 2 + 0] = i0; topi[n * 2 + 1] = i1;
    topv[n * 2 + 0] = v0; topv[n * 2 + 1] = v1;
    for (int e = 0; e < NEXP; ++e) { sMe[wid][e] = p[e]; sCe[wid][e] = (e == i0) ? 1.f : 0.f; }
  }
  __syncthreads();
  if (tid < NEXP) {
    float m = 0.f, c = 0.f;
    for (int w = 0; w < 8; ++w) { m += sMe[w][tid]; c += sCe[w][tid]; }
    me_part[blockIdx.x * NEXP + tid] = m;
    ce_part[blockIdx.x * NEXP + tid] = c;
  }
}

// ------------- scan: deterministic slot-major positions + l_aux --------------
__global__ void scan_kernel(const int* __restrict__ topi,
                            int* __restrict__ flat_idx, float* __restrict__ wgt,
                            int* __restrict__ slot_token,
                            const float* __restrict__ me_part, const float* __restrict__ ce_part,
                            float* __restrict__ laux_out) {
  __shared__ int hist[256][NEXP];
  __shared__ float sred[NEXP];
  const int t = threadIdx.x;
  const int CHUNK = (2 * NTOK) / 256;  // 64 assignments per thread

  for (int i = 0; i < (CAP * NEXP) / 256; ++i)
    slot_token[t + i * 256] = -1;

  int h[NEXP];
#pragma unroll
  for (int e = 0; e < NEXP; ++e) h[e] = 0;
  const int a0 = t * CHUNK;
  for (int i = 0; i < CHUNK; ++i) {
    int a = a0 + i;
    int slot = a / NTOK, n = a % NTOK;
    h[topi[n * 2 + slot]]++;
  }
#pragma unroll
  for (int e = 0; e < NEXP; ++e) hist[t][e] = h[e];
  __syncthreads();

  int base[NEXP];
#pragma unroll
  for (int e = 0; e < NEXP; ++e) base[e] = 0;
  for (int tt = 0; tt < t; ++tt)
#pragma unroll
    for (int e = 0; e < NEXP; ++e) base[e] += hist[tt][e];

  for (int i = 0; i < CHUNK; ++i) {
    int a = a0 + i;
    int slot = a / NTOK, n = a % NTOK;
    int e = topi[n * 2 + slot];
    int pos = base[e]++;
    bool valid = pos < CAP;
    int flat = e * CAP + (valid ? pos : 0);
    flat_idx[n * 2 + slot] = flat;
    wgt[n * 2 + slot] = valid ? 1.f : 0.f;
    if (valid) slot_token[flat] = n;
  }

  if (t < NEXP) {
    float m = 0.f, c = 0.f;
    for (int b = 0; b < NTOK / 8; ++b) { m += me_part[b * NEXP + t]; c += ce_part[b * NEXP + t]; }
    sred[t] = (m / (float)NTOK) * (c / (float)NTOK);
  }
  __syncthreads();
  if (t == 0) {
    float s = 0.f;
    for (int e = 0; e < NEXP; ++e) s += sred[e];
    laux_out[0] = (float)NEXP * s;
  }
}

// ------- weight convert+transpose: dst[e][c][r] = bf16(src[e][r][c]) ---------
__global__ void transpose_bf16_kernel(const float* __restrict__ src,
                                      unsigned short* __restrict__ dst, int R, int C) {
  __shared__ float tile[32][33];
  const int e = blockIdx.z;
  const int c0 = blockIdx.x * 32, r0 = blockIdx.y * 32;
  const int lane = threadIdx.x & 31, rr = threadIdx.x >> 5;
  const float* s = src + (size_t)e * R * C;
  unsigned short* d = dst + (size_t)e * R * C;
#pragma unroll
  for (int p = 0; p < 4; ++p) {
    int r = rr + p * 8;
    tile[r][lane] = s[(size_t)(r0 + r) * C + c0 + lane];
  }
  __syncthreads();
#pragma unroll
  for (int p = 0; p < 4; ++p) {
    int cc = rr + p * 8;
    d[(size_t)(c0 + cc) * R + r0 + lane] = f2bf(tile[lane][cc]);
  }
}

// --------------- dispatch: build bf16 [E*CAP, D] buffer (full write) ---------
__global__ void dispatch_kernel(const float* __restrict__ x, const int* __restrict__ slot_token,
                                unsigned short* __restrict__ Xd) {
  const int wid = threadIdx.x >> 5, lane = threadIdx.x & 31;
  const int row = blockIdx.x * 8 + wid;
  const int tok = slot_token[row];
  unsigned short* o = Xd + (size_t)row * DDIM;
  if (tok >= 0) {
    const float* xr = x + (size_t)tok * DDIM;
    for (int it = 0; it < DDIM / 32; ++it) o[it * 32 + lane] = f2bf(xr[it * 32 + lane]);
  } else {
    for (int it = 0; it < DDIM / 32; ++it) o[it * 32 + lane] = 0;
  }
}

// --------------------- tiled bf16 WMMA GEMM (per expert) --------------------
// A: [M][K] bf16 row-major.  Bt: [N][K] bf16 (pre-transposed).  C = A*B + bias.
// Block tile 128x128, BK=32; 8 waves, each 32(M) x 64(N) = 2x4 wmma accumulators.
template <bool RELU_BF16_OUT>
__global__ __launch_bounds__(256) void moe_gemm_kernel(
    const unsigned short* __restrict__ Ag, const unsigned short* __restrict__ Btg,
    const float* __restrict__ biasg, void* __restrict__ Cg, int M, int N, int K) {
  __shared__ unsigned short As[128 * 32];
  __shared__ unsigned short Bs[128 * 32];

  const int e = blockIdx.z;
  const unsigned short* A  = Ag  + (size_t)e * M * K;
  const unsigned short* Bt = Btg + (size_t)e * N * K;
  const float* bias = biasg + (size_t)e * N;

  const int m0 = blockIdx.y * 128;
  const int n0 = blockIdx.x * 128;
  const int tid = threadIdx.x;
  const int wid = tid >> 5, lane = tid & 31;
  const int lr = lane & 15, hi = lane >> 4;
  const int wm = (wid & 3) * 32;   // wave M offset
  const int wn = (wid >> 2) * 64;  // wave N offset

  v8f acc[2][4];
#pragma unroll
  for (int i = 0; i < 2; ++i)
#pragma unroll
    for (int j = 0; j < 4; ++j)
#pragma unroll
      for (int r = 0; r < 8; ++r) acc[i][j][r] = 0.f;

  const int lrow = tid >> 1;        // 0..127
  const int lko  = (tid & 1) * 16;  // 0 or 16 (ushorts)

  for (int kt = 0; kt < K; kt += 32) {
    // global -> LDS (contiguous 32B per thread per matrix)
    const uint4* ga = reinterpret_cast<const uint4*>(A  + (size_t)(m0 + lrow) * K + kt + lko);
    const uint4* gb = reinterpret_cast<const uint4*>(Bt + (size_t)(n0 + lrow) * K + kt + lko);
    uint4 va0 = ga[0], va1 = ga[1];
    uint4 vb0 = gb[0], vb1 = gb[1];
    uint4* la = reinterpret_cast<uint4*>(As + lrow * 32 + lko);
    uint4* lb = reinterpret_cast<uint4*>(Bs + lrow * 32 + lko);
    la[0] = va0; la[1] = va1;
    lb[0] = vb0; lb[1] = vb1;
    __syncthreads();

    // fragments per CDNA5 bf16 WMMA VGPR layout
    Frag16 af[2], bfr[4];
#pragma unroll
    for (int i = 0; i < 2; ++i) {
      const uint4* pa = reinterpret_cast<const uint4*>(As + (wm + i * 16 + lr) * 32 + hi * 8);
      af[i].q[0] = pa[0];      // K = c..c+7   (c = hi*8)
      af[i].q[1] = pa[2];      // K = c+16..c+23
    }
#pragma unroll
    for (int j = 0; j < 4; ++j) {
      const uint4* pb = reinterpret_cast<const uint4*>(Bs + (wn + j * 16 + lr) * 32 + hi * 8);
      bfr[j].q[0] = pb[0];     // K = c..c+7   (c = hi*8)
      bfr[j].q[1] = pb[2];     // K = c+16..c+23
    }
#pragma unroll
    for (int i = 0; i < 2; ++i)
#pragma unroll
      for (int j = 0; j < 4; ++j)
        acc[i][j] = __builtin_amdgcn_wmma_f32_16x16x32_bf16(
            false, af[i].v, false, bfr[j].v, (short)0, acc[i][j], false, false);
    __syncthreads();
  }

  // epilogue: bias (+ReLU -> bf16) or f32 store; C/D layout: lane lr = col,
  // VGPR r -> row r + hi*8
#pragma unroll
  for (int j = 0; j < 4; ++j) {
    int col = n0 + wn + j * 16 + lr;
    float bv = bias[col];
#pragma unroll
    for (int i = 0; i < 2; ++i) {
#pragma unroll
      for (int r = 0; r < 8; ++r) {
        int row = m0 + wm + i * 16 + hi * 8 + r;
        float v = acc[i][j][r] + bv;
        if (RELU_BF16_OUT) {
          v = fmaxf(v, 0.f);
          ((unsigned short*)Cg)[(size_t)e * M * N + (size_t)row * N + col] = f2bf(v);
        } else {
          ((float*)Cg)[(size_t)e * M * N + (size_t)row * N + col] = v;
        }
      }
    }
  }
}

// ------------------- combine: out[n] = sum_slot gate*Y[flat] -----------------
__global__ void combine_kernel(const float* __restrict__ Y, const int* __restrict__ flat_idx,
                               const float* __restrict__ topv, const float* __restrict__ wgt,
                               float* __restrict__ out) {
  const int wid = threadIdx.x >> 5, lane = threadIdx.x & 31;
  const int n = blockIdx.x * 8 + wid;
  const int f0 = flat_idx[n * 2 + 0], f1 = flat_idx[n * 2 + 1];
  const float c0 = topv[n * 2 + 0] * wgt[n * 2 + 0];
  const float c1 = topv[n * 2 + 1] * wgt[n * 2 + 1];
  const float* y0 = Y + (size_t)f0 * DDIM;
  const float* y1 = Y + (size_t)f1 * DDIM;
  float* o = out + (size_t)n * DDIM;
  for (int it = 0; it < DDIM / 32; ++it) {
    int d = it * 32 + lane;
    o[d] = c0 * y0[d] + c1 * y1[d];
  }
}

extern "C" void kernel_launch(void* const* d_in, const int* in_sizes, int n_in,
                              void* d_out, int out_size, void* d_ws, size_t ws_size,
                              hipStream_t stream) {
  (void)in_sizes; (void)n_in; (void)out_size; (void)ws_size;
  const float* x     = (const float*)d_in[0];
  const float* wg    = (const float*)d_in[1];
  const float* fc1_w = (const float*)d_in[2];
  const float* fc1_b = (const float*)d_in[3];
  const float* fc2_w = (const float*)d_in[4];
  const float* fc2_b = (const float*)d_in[5];
  float* out = (float*)d_out;

  char* ws = (char*)d_ws;
  const size_t MB = 1024ull * 1024ull;
  unsigned short* w1t = (unsigned short*)(ws);              // 64 MB  [E][H][D] bf16
  unsigned short* w2t = (unsigned short*)(ws + 64 * MB);    // 64 MB  [E][D][H] bf16
  unsigned short* Xd  = (unsigned short*)(ws + 128 * MB);   // 32 MB  [E*CAP][D] bf16
  unsigned short* Hb  = (unsigned short*)(ws + 160 * MB);   // 128 MB [E*CAP][H] bf16
  char* sm = ws + 288 * MB;
  int*   topi = (int*)(sm);
  float* topv = (float*)(sm + 1 * 65536);
  int*   flat = (int*)(sm + 2 * 65536);
  float* wgt  = (float*)(sm + 3 * 65536);
  int*   slot = (int*)(sm + 4 * 65536);
  float* mep  = (float*)(sm + 5 * 65536);
  float* cep  = (float*)(sm + 5 * 65536 + 32768);
  float* Y    = (float*)(ws);  // 64 MB, overlays w1t (dead after GEMM1)

  gate_kernel<<<NTOK / 8, 256, 0, stream>>>(x, wg, topi, topv, mep, cep);
  scan_kernel<<<1, 256, 0, stream>>>(topi, flat, wgt, slot, mep, cep,
                                     out + (size_t)NTOK * DDIM);
  transpose_bf16_kernel<<<dim3(HDIM / 32, DDIM / 32, NEXP), 256, 0, stream>>>(
      fc1_w, w1t, DDIM, HDIM);
  transpose_bf16_kernel<<<dim3(DDIM / 32, HDIM / 32, NEXP), 256, 0, stream>>>(
      fc2_w, w2t, HDIM, DDIM);
  dispatch_kernel<<<(NEXP * CAP) / 8, 256, 0, stream>>>(x, slot, Xd);
  moe_gemm_kernel<true><<<dim3(HDIM / 128, CAP / 128, NEXP), 256, 0, stream>>>(
      Xd, w1t, fc1_b, Hb, CAP, HDIM, DDIM);
  moe_gemm_kernel<false><<<dim3(DDIM / 128, CAP / 128, NEXP), 256, 0, stream>>>(
      Hb, w2t, fc2_b, Y, CAP, DDIM, HDIM);
  combine_kernel<<<NTOK / 8, 256, 0, stream>>>(Y, flat, topv, wgt, out);
}
